// WindowedMultiheadAttention_75574244540550
// MI455X (gfx1250) — matrix operand order repaired and checked by reading, not compiled, not run on hardware
//
#include <hip/hip_runtime.h>

// ---------------------------------------------------------------------------
// Windowed multi-head attention for MI455X (gfx1250, wave32, WMMA).
// All matmuls use v_wmma_f32_16x16x32_bf16 (f32 accumulate); f32->bf16 via
// native v_cvt_pk_bf16_f32; GEMMs use 2x2 register tiling for operand reuse;
// WGP-scope global prefetch of the next K-slice.
// ---------------------------------------------------------------------------

typedef __attribute__((ext_vector_type(16))) __bf16 v16bf;
typedef __attribute__((ext_vector_type(8)))  __bf16 v8bf;
typedef __attribute__((ext_vector_type(8)))  float  v8f;

#define WMMA_BF16(a, b, c)                                                     \
  __builtin_amdgcn_wmma_f32_16x16x32_bf16(false, (a), false, (b), (short)0,    \
                                          (c), false, false)

__device__ __forceinline__ void ld8f(const float* p, float* o) {
  float4 a = *reinterpret_cast<const float4*>(p);
  float4 b = *reinterpret_cast<const float4*>(p + 4);
  o[0] = a.x; o[1] = a.y; o[2] = a.z; o[3] = a.w;
  o[4] = b.x; o[5] = b.y; o[6] = b.z; o[7] = b.w;
}

__device__ __forceinline__ v16bf cvt16(const float* t) {
  v16bf r;
#pragma unroll
  for (int j = 0; j < 16; ++j) r[j] = (__bf16)t[j];  // v_cvt_pk_bf16_f32
  return r;
}

// A-operand (16x32 bf16) from an f32 row. Lane layout (contiguous):
// lanes 0-15: K = 0..15; lanes 16-31: K = 16..31.
// 'lo' = (lane>>4)*8. rowp points at this lane's row at the K-slice base.
__device__ __forceinline__ v16bf ldA_f32(const float* rowp, int lo) {
  float t[16];
  ld8f(rowp + 2 * lo, t);
  ld8f(rowp + 2 * lo + 8, t + 8);
  return cvt16(t);
}

// B-operand (32x16 bf16) from an f32 row: lane holds column c = lane&15,
// K = (lane>>4)*16 + 0..15, contiguous. p points at the 16-element run.
__device__ __forceinline__ v16bf ldB_f32(const float* p) {
  float t[16];
  ld8f(p, t);
  ld8f(p + 8, t + 8);
  return cvt16(t);
}

// Same operand loads straight from bf16 memory (global workspace or LDS).
__device__ __forceinline__ v16bf ldA_b16(const __bf16* rowp, int lo) {
  v8bf a = *reinterpret_cast<const v8bf*>(rowp + 2 * lo);
  v8bf b = *reinterpret_cast<const v8bf*>(rowp + 2 * lo + 8);
  v16bf u;
#pragma unroll
  for (int j = 0; j < 8; ++j) { u[j] = a[j]; u[8 + j] = b[j]; }
  return u;
}

__device__ __forceinline__ v16bf ldB_b16(const __bf16* p) {
  return *reinterpret_cast<const v16bf*>(p);
}

// ---------------------------------------------------------------------------
// Kernel 1: qkv = x @ qkv_w^T + qkv_b  (8192x256 @ 256x768)
// One wave per 32x32 output block (2x2 tiles of 16x16); K=256 -> 32 WMMAs.
// q scaled by E^-0.5 -> qb[B,H,N,32]; k -> kb[B,H,N,32]; v -> vT[B,H,32,N].
// ---------------------------------------------------------------------------
__global__ __launch_bounds__(128) void qkv_kernel(
    const float* __restrict__ x, const float* __restrict__ qkv_w,
    const float* __restrict__ qkv_b, __bf16* __restrict__ qb,
    __bf16* __restrict__ kb, __bf16* __restrict__ vT) {
  const int lane = threadIdx.x & 31;
  const int wg   = blockIdx.x * 4 + (threadIdx.x >> 5);  // 0..6143
  const int cg = wg % 24, rg = wg / 24;   // 24 col groups x 256 row groups
  const int c00 = cg * 32;                // 32-col block: one (which, head)
  const int r00 = rg * 32;                // 32-row block: one batch
  const int which = c00 >> 8;             // 0=q,1=k,2=v (uniform)
  const int head  = (c00 >> 5) & 7;       // uniform within the block
  const int lh = lane >> 4, ll = lane & 15, lo = lh * 8;

  const float* xr0 = x + (r00 + ll) * 256;
  const float* xr1 = xr0 + 16 * 256;
  const float* wr0 = qkv_w + (c00 + ll) * 256 + lh * 16;
  const float* wr1 = wr0 + 16 * 256;

  v8f a00 = {0.f,0.f,0.f,0.f,0.f,0.f,0.f,0.f}, a01 = a00, a10 = a00, a11 = a00;
#pragma unroll
  for (int kk = 0; kk < 8; ++kk) {
    if (kk < 7) {  // WGP-scope prefetch of next K-slice (global_prefetch_b8)
      __builtin_prefetch(xr0 + (kk + 1) * 32, 0, 3);
      __builtin_prefetch(wr0 + (kk + 1) * 32, 0, 3);
    }
    const v16bf A0 = ldA_f32(xr0 + kk * 32, lo);
    const v16bf A1 = ldA_f32(xr1 + kk * 32, lo);
    const v16bf B0 = ldB_f32(wr0 + kk * 32);
    const v16bf B1 = ldB_f32(wr1 + kk * 32);
    a00 = WMMA_BF16(A0, B0, a00);
    a01 = WMMA_BF16(A0, B1, a01);
    a10 = WMMA_BF16(A1, B0, a10);
    a11 = WMMA_BF16(A1, B1, a11);
  }

  const int bidx = r00 >> 10;             // batch
  const int bh   = bidx * 8 + head;
  const float bias0 = qkv_b[c00 + ll];
  const float bias1 = qkv_b[c00 + 16 + ll];
#pragma unroll
  for (int sr = 0; sr < 2; ++sr) {
#pragma unroll
    for (int sc = 0; sc < 2; ++sc) {
      const v8f& acc = sr == 0 ? (sc == 0 ? a00 : a01) : (sc == 0 ? a10 : a11);
      const float bias = sc == 0 ? bias0 : bias1;
      const int d = sc * 16 + ll;         // feature within head
#pragma unroll
      for (int r = 0; r < 8; ++r) {
        const int n = (r00 & 1023) + sr * 16 + r + lh * 8;  // C-layout row
        const float v = acc[r] + bias;
        if (which == 0)      qb[(bh * 1024 + n) * 32 + d] = (__bf16)(v * 0.0625f);
        else if (which == 1) kb[(bh * 1024 + n) * 32 + d] = (__bf16)v;
        else                 vT[(bh * 32 + d) * 1024 + n] = (__bf16)v;
      }
    }
  }
}

// ---------------------------------------------------------------------------
// Kernel 2: windowed attention. One wave per (b, head, 16-query tile).
// Query tile shares grid-row h -> only 7 key grid-rows (14 key tiles),
// exploiting the 7x11 window structurally (224 keys instead of 1024).
// QK^T: 14 WMMAs (K = d_h = 32). Softmax in f32 with shfl_xor row
// reductions. P staged through LDS (bf16) to build the A-operand for AV:
// 14 more WMMAs.
// ---------------------------------------------------------------------------
__global__ __launch_bounds__(128) void attn_kernel(
    const __bf16* __restrict__ qb, const __bf16* __restrict__ kb,
    const __bf16* __restrict__ vT, __bf16* __restrict__ ob) {
  __shared__ __align__(16) __bf16 Pb[4][16][224];

  const int lane = threadIdx.x & 31;
  const int warp = threadIdx.x >> 5;
  const int wg   = blockIdx.x * 4 + warp;      // 0..4095
  const int qt = wg & 63, h = (wg >> 6) & 7, b = wg >> 9;
  const int bh = b * 8 + h;
  const int n0 = qt * 16;
  const int hq  = n0 >> 5;     // grid row of all queries in this tile
  const int wq0 = n0 & 31;     // 0 or 16
  const int lh = lane >> 4, ll = lane & 15, lo = lh * 8;

  // Q A-operand, K = d_h = 32 (one load, reused for all 14 score WMMAs)
  const v16bf Aq = ldA_b16(qb + (bh * 1024 + n0 + ll) * 32, lo);

  v8f S[14];
#pragma unroll
  for (int i = 0; i < 7; ++i) {
    const int hp = hq - 3 + i;                  // key grid row (uniform)
    const bool vh = (hp >= 0) && (hp < 32);
#pragma unroll
    for (int mt = 0; mt < 2; ++mt) {
      v8f s = {0.f, 0.f, 0.f, 0.f, 0.f, 0.f, 0.f, 0.f};
      if (vh) {   // wave-uniform branch: EXEC stays all-ones for the WMMA
        const __bf16* kr =
            kb + (bh * 1024 + hp * 32 + mt * 16 + ll) * 32 + lh * 16;
        s = WMMA_BF16(Aq, ldB_b16(kr), s);
      }
#pragma unroll
      for (int r = 0; r < 8; ++r) {
        const int dw = (wq0 + r + lh * 8) - (mt * 16 + ll);  // w_q - w_k
        const int adw = dw < 0 ? -dw : dw;
        S[i * 2 + mt][r] = (vh && adw <= 5) ? s[r] : -1e30f;
      }
    }
  }

  // Row softmax. C layout: row = r + 8*lh, col = lane&15 -> reduce over
  // the 16-lane half-group (xor offsets 1,2,4,8 stay within the group).
  float inv[8];
#pragma unroll
  for (int r = 0; r < 8; ++r) {
    float mx = S[0][r];
#pragma unroll
    for (int t = 1; t < 14; ++t) mx = fmaxf(mx, S[t][r]);
#pragma unroll
    for (int off = 1; off < 16; off <<= 1)
      mx = fmaxf(mx, __shfl_xor(mx, off, 32));
    float sum = 0.f;
#pragma unroll
    for (int t = 0; t < 14; ++t) {
      const float p = __expf(S[t][r] - mx);
      S[t][r] = p;
      sum += p;
    }
#pragma unroll
    for (int off = 1; off < 16; off <<= 1) sum += __shfl_xor(sum, off, 32);
    inv[r] = __builtin_amdgcn_rcpf(sum);
  }

  // P -> LDS (bf16, row-major 16x224), per-wave private region.
#pragma unroll
  for (int t = 0; t < 14; ++t)
#pragma unroll
    for (int r = 0; r < 8; ++r)
      Pb[warp][r + lh * 8][t * 16 + ll] = (__bf16)(S[t][r] * inv[r]);

  // AV: out(16x32) = P(16x224) @ V(224x32); 7 K-chunks x 2 d-column tiles.
  v8f o0 = {0.f, 0.f, 0.f, 0.f, 0.f, 0.f, 0.f, 0.f};
  v8f o1 = {0.f, 0.f, 0.f, 0.f, 0.f, 0.f, 0.f, 0.f};
#pragma unroll
  for (int i = 0; i < 7; ++i) {
    int hp = hq - 3 + i;
    const int hpc = hp < 0 ? 0 : (hp > 31 ? 31 : hp);  // clamp; P rows are 0
    const v16bf Ap = ldA_b16(&Pb[warp][ll][i * 32], lo);  // ds loads
    const __bf16* v0 = vT + (bh * 32 + 0 + ll) * 1024 + hpc * 32 + lh * 16;
    const __bf16* v1 = vT + (bh * 32 + 16 + ll) * 1024 + hpc * 32 + lh * 16;
    o0 = WMMA_BF16(Ap, ldB_b16(v0), o0);
    o1 = WMMA_BF16(Ap, ldB_b16(v1), o1);
  }

  // ob[B*N, 256] bf16; column = head*32 + d
#pragma unroll
  for (int r = 0; r < 8; ++r) {
    const int row = (b * 1024 + n0 + r + lh * 8) * 256 + h * 32;
    ob[row + ll]      = (__bf16)o0[r];
    ob[row + 16 + ll] = (__bf16)o1[r];
  }
}

// ---------------------------------------------------------------------------
// Kernel 3: y = ob @ proj_w^T + proj_b  (8192x256 @ 256x256), f32 output.
// One wave per 32x32 output block (2x2 tiles); K=256 -> 32 WMMAs.
// ---------------------------------------------------------------------------
__global__ __launch_bounds__(128) void proj_kernel(
    const __bf16* __restrict__ ob, const float* __restrict__ proj_w,
    const float* __restrict__ proj_b, float* __restrict__ out) {
  const int lane = threadIdx.x & 31;
  const int wg   = blockIdx.x * 4 + (threadIdx.x >> 5);  // 0..2047
  const int cg = wg & 7, rg = wg >> 3;   // 8 col groups x 256 row groups
  const int c00 = cg * 32, r00 = rg * 32;
  const int lh = lane >> 4, ll = lane & 15, lo = lh * 8;

  const __bf16* ar0 = ob + (r00 + ll) * 256;
  const __bf16* ar1 = ar0 + 16 * 256;
  const float* wr0 = proj_w + (c00 + ll) * 256 + lh * 16;
  const float* wr1 = wr0 + 16 * 256;

  v8f a00 = {0.f,0.f,0.f,0.f,0.f,0.f,0.f,0.f}, a01 = a00, a10 = a00, a11 = a00;
#pragma unroll
  for (int kk = 0; kk < 8; ++kk) {
    if (kk < 7) {  // WGP-scope prefetch of next K-slice
      __builtin_prefetch(ar0 + (kk + 1) * 32, 0, 3);
      __builtin_prefetch(wr0 + (kk + 1) * 32, 0, 3);
    }
    const v16bf A0 = ldA_b16(ar0 + kk * 32, lo);
    const v16bf A1 = ldA_b16(ar1 + kk * 32, lo);
    const v16bf B0 = ldB_f32(wr0 + kk * 32);
    const v16bf B1 = ldB_f32(wr1 + kk * 32);
    a00 = WMMA_BF16(A0, B0, a00);
    a01 = WMMA_BF16(A0, B1, a01);
    a10 = WMMA_BF16(A1, B0, a10);
    a11 = WMMA_BF16(A1, B1, a11);
  }

  const float bias0 = proj_b[c00 + ll];
  const float bias1 = proj_b[c00 + 16 + ll];
#pragma unroll
  for (int sr = 0; sr < 2; ++sr) {
#pragma unroll
    for (int sc = 0; sc < 2; ++sc) {
      const v8f& acc = sr == 0 ? (sc == 0 ? a00 : a01) : (sc == 0 ? a10 : a11);
      const float bias = sc == 0 ? bias0 : bias1;
#pragma unroll
      for (int r = 0; r < 8; ++r)
        out[(r00 + sr * 16 + r + lh * 8) * 256 + c00 + sc * 16 + ll] =
            acc[r] + bias;
    }
  }
}

// ---------------------------------------------------------------------------
extern "C" void kernel_launch(void* const* d_in, const int* in_sizes, int n_in,
                              void* d_out, int out_size, void* d_ws,
                              size_t ws_size, hipStream_t stream) {
  (void)in_sizes; (void)n_in; (void)out_size; (void)ws_size;
  const float* x      = (const float*)d_in[0];  // [8,1024,256]
  const float* qkv_w  = (const float*)d_in[1];  // [768,256]
  const float* qkv_b  = (const float*)d_in[2];  // [768]
  const float* proj_w = (const float*)d_in[3];  // [256,256]
  const float* proj_b = (const float*)d_in[4];  // [256]

  // Workspace: 4 bf16 regions of 2Mi elements each = 16 MiB total.
  __bf16* qb = (__bf16*)d_ws;
  __bf16* kb = qb + (size_t)2 * 1024 * 1024;
  __bf16* vT = kb + (size_t)2 * 1024 * 1024;
  __bf16* ob = vT + (size_t)2 * 1024 * 1024;

  qkv_kernel<<<1536, 128, 0, stream>>>(x, qkv_w, qkv_b, qb, kb, vT);
  attn_kernel<<<1024, 128, 0, stream>>>(qb, kb, vT, ob);
  proj_kernel<<<512, 128, 0, stream>>>(ob, proj_w, proj_b, (float*)d_out);
}
